// MultiModalMambaKANEncoderMask_19164144075113
// MI455X (gfx1250) — matrix operand order repaired and checked by reading, not compiled, not run on hardware
//
#include <hip/hip_runtime.h>
#include <hip/hip_bf16.h>

typedef __bf16 bf16_t;
typedef __attribute__((ext_vector_type(16))) __bf16 v16bf;
typedef __attribute__((ext_vector_type(8)))  __bf16 v8bf;
typedef __attribute__((ext_vector_type(8)))  float  v8f;

// ---- problem constants ----
static const int NB      = 8;      // batch
static const int TPRE    = 6144;
static const int CIN     = 8;
static const int TOK     = 512;    // tokens after patching (8192/16)
static const int MASK0   = 384;    // first masked token (6144/16)
static const int DM      = 256;    // d_model
static const int DI      = 512;    // d_inner
static const int NH      = 8;      // heads
static const int HD      = 64;     // head dim
static const int DSTATE  = 32;
static const int CDIM    = 576;    // conv dim
static const int DPROJ   = 1096;
static const int DPROJP  = 1152;   // padded to multiple of 64
static const int ROWS    = NB * TOK;       // 4096
static const int RPATCH  = DM * 16;        // 4096 reduction dim of patch conv
static const int OUTC    = 8;
static const int OUTLEN  = 2048;

__device__ __forceinline__ float siluf(float x) { return x / (1.f + __expf(-x)); }
__device__ __forceinline__ float softplusf(float x) {
  return (x > 20.f) ? x : logf(1.f + __expf(x));
}
__device__ __forceinline__ bf16_t f2bf(float f) {
  unsigned int u = __float_as_uint(f);
  unsigned int r = (u + 0x7fffu + ((u >> 16) & 1u)) >> 16;
  unsigned short s = (unsigned short)r;
  bf16_t b;
  __builtin_memcpy(&b, &s, 2);
  return b;
}

// ============================================================================
// bf16 WMMA GEMM: C[M,N] (f32) = A[M,K](bf16 row-major) * BT[Npad,K]^T
// One wave per 16x64 tile: the A fragment is loaded once per K-step and
// reused by 4 independent accumulator chains (4 WMMAs / K-step).
// Optional accumulate-into-C (residual) and bias add at store.
// Rows of BT beyond N must be zero-padded up to a multiple of 64.
// ============================================================================
__global__ void gemm_bf16_wmma(const bf16_t* __restrict__ A,
                               const bf16_t* __restrict__ BT,
                               float* __restrict__ C,
                               int M, int N, int K, int ldc,
                               const float* __restrict__ bias,
                               int accumulate) {
  const int lane = threadIdx.x;           // 0..31
  const int half = lane >> 4;             // 0/1
  const int l16  = lane & 15;
  const int m0 = blockIdx.y << 4;
  const int n0 = blockIdx.x << 6;         // 64-wide tile
  const long aoff = (long)(m0 + l16) * K; // A row for this lane
  long boff[4];
#pragma unroll
  for (int s = 0; s < 4; ++s) boff[s] = (long)(n0 + s * 16 + l16) * K;
  const int ka = half << 3;               // A K-chunk base: 0 or 8
  const int kb = half << 4;               // unused: B loads also use ka

  v8f acc[4];
#pragma unroll
  for (int s = 0; s < 4; ++s) {
    if (accumulate) {
#pragma unroll
      for (int r = 0; r < 8; ++r) {
        const int mm = m0 + r + (half << 3);
        const int nn = n0 + s * 16 + l16;
        acc[s][r] = (nn < N) ? C[(long)mm * ldc + nn] : 0.f;
      }
    } else {
#pragma unroll
      for (int r = 0; r < 8; ++r) acc[s][r] = 0.f;
    }
  }

#pragma unroll 2
  for (int k = 0; k < K; k += 32) {
    // A 16x32 bf16 fragment: lanes<16 hold K {0..7,16..23}; lanes>=16 {8..15,24..31}
    const v8bf a0 = *(const v8bf*)(A + aoff + k + ka);
    const v8bf a1 = *(const v8bf*)(A + aoff + k + 16 + ka);
    const v16bf av = __builtin_shufflevector(a0, a1, 0,1,2,3,4,5,6,7,8,9,10,11,12,13,14,15);
#pragma unroll
    for (int s = 0; s < 4; ++s) {
      // B 32x16 bf16 fragment: lanes<16 hold K {0..7,16..23}; lanes>=16 {8..15,24..31}
      const v8bf b0 = *(const v8bf*)(BT + boff[s] + k + ka);
      const v8bf b1 = *(const v8bf*)(BT + boff[s] + k + 16 + ka);
      const v16bf bv = __builtin_shufflevector(b0, b1, 0,1,2,3,4,5,6,7,8,9,10,11,12,13,14,15);
      acc[s] = __builtin_amdgcn_wmma_f32_16x16x32_bf16(false, av, false, bv,
                                                       (short)0, acc[s], false, false);
    }
  }

#pragma unroll
  for (int s = 0; s < 4; ++s) {
    const int nn = n0 + s * 16 + l16;
    if (nn < N) {
      const float bv = bias ? bias[nn] : 0.f;
#pragma unroll
      for (int r = 0; r < 8; ++r) {
        const int mm = m0 + r + (half << 3);
        C[(long)mm * ldc + nn] = acc[s][r] + bv;
      }
    }
  }
}

// ============================================================================
// Weight prep: f32 -> bf16 (plain, and transposed K x N -> Npad x K with pad)
// ============================================================================
__global__ void cvt_bf(const float* __restrict__ src, bf16_t* __restrict__ dst, long n) {
  long i = (long)blockIdx.x * blockDim.x + threadIdx.x;
  if (i < n) dst[i] = f2bf(src[i]);
}

__global__ void cvtT_bf(const float* __restrict__ src, bf16_t* __restrict__ dst,
                        int K, int N, int Npad) {
  long i = (long)blockIdx.x * blockDim.x + threadIdx.x;
  if (i >= (long)Npad * K) return;
  const int k = (int)(i % K);
  const long n = i / K;
  dst[i] = (n < N) ? f2bf(src[(long)k * N + n]) : f2bf(0.f);
}

// ============================================================================
// Fused pre-conv (C_IN->D_MODEL, K=5, same-pad) + SiLU, laid out as the
// A-matrix of the patch GEMM: A[bt][ic*16+kk] = silu(preconv[ic][t*16+kk])
// ============================================================================
__global__ void build_patch_A(const float* __restrict__ xp,
                              const float* __restrict__ w,
                              const float* __restrict__ bias,
                              bf16_t* __restrict__ A) {
  long idx = (long)blockIdx.x * 256 + threadIdx.x;   // ROWS * RPATCH
  const int r = (int)(idx & 4095);
  const long bt = idx >> 12;
  const int ic = r >> 4, kk = r & 15;
  const int t = (int)(bt & 511);
  const long b = bt >> 9;
  const int tau = t * 16 + kk;
  float acc = bias[ic];
#pragma unroll
  for (int k5 = 0; k5 < 5; ++k5) {
    const int ts = tau + k5 - 2;
    if (ts >= 0 && ts < TPRE) {
      const float* xr = xp + (b * TPRE + ts) * CIN;
#pragma unroll
      for (int j = 0; j < CIN; ++j) acc += w[ic * 40 + j * 5 + k5] * xr[j];
    }
  }
  A[idx] = f2bf(siluf(acc));
}

// aux_bias[b][c] = silu(y_aux[b] @ aux_w + aux_b)
__global__ void aux_bias_k(const float* __restrict__ y_aux,
                           const float* __restrict__ aux_w,
                           const float* __restrict__ aux_b,
                           float* __restrict__ out) {
  const int c = threadIdx.x, b = blockIdx.x;
  float acc = aux_b[c];
#pragma unroll
  for (int i = 0; i < 16; ++i) acc += y_aux[b * 16 + i] * aux_w[i * DM + c];
  out[b * DM + c] = siluf(acc);
}

// tokens: mask + patch bias + aux bias; write forward and time-reversed copies
__global__ void assemble_tokens(const float* __restrict__ xtok,
                                const float* __restrict__ patch_b,
                                const float* __restrict__ mask_token,
                                const float* __restrict__ auxb,
                                float* __restrict__ hf, float* __restrict__ hb) {
  long idx = (long)blockIdx.x * 256 + threadIdx.x;   // ROWS * DM
  const int c = (int)(idx & 255);
  const long bt = idx >> 8;
  const int t = (int)(bt & 511);
  const long b = bt >> 9;
  float v = (t >= MASK0) ? mask_token[c] : (xtok[idx] + patch_b[c]);
  v += auxb[b * DM + c];
  hf[idx] = v;
  hb[((b << 9) + (TOK - 1 - t)) * DM + c] = v;
}

// rmsnorm over D=256, output bf16 (GEMM A operand)
__global__ void rmsnorm256_bf(const float* __restrict__ x,
                              const float* __restrict__ w,
                              bf16_t* __restrict__ out) {
  __shared__ float red[256];
  const int row = blockIdx.x, c = threadIdx.x;
  const float v = x[(long)row * DM + c];
  red[c] = v * v;
  __syncthreads();
  for (int s = 128; s > 0; s >>= 1) { if (c < s) red[c] += red[c + s]; __syncthreads(); }
  const float scale = rsqrtf(red[0] * (1.f / DM) + 1e-5f);
  out[(long)row * DM + c] = f2bf(v * scale * w[c]);
}

// depthwise causal conv (K=4) over the xBC slice of zxbcdt, + SiLU
__global__ void dwconv_silu(const float* __restrict__ zx,
                            const float* __restrict__ cw,
                            const float* __restrict__ cb,
                            float* __restrict__ xBC) {
  long idx = (long)blockIdx.x * 256 + threadIdx.x;   // ROWS * CDIM
  const int c = (int)(idx % CDIM);
  const long bt = idx / CDIM;
  const int t = (int)(bt & 511);
  const long b = bt >> 9;
  float acc = cb[c];
#pragma unroll
  for (int k = 0; k < 4; ++k) {
    const int ts = t - 3 + k;
    if (ts >= 0) acc += cw[c * 4 + k] * zx[((b << 9) + ts) * (long)DPROJ + DI + c];
  }
  xBC[idx] = siluf(acc);
}

// sequential SSM scan; one block per (head, batch), 64 threads = HEADDIM
__global__ void ssm_scan(const float* __restrict__ zx,
                         const float* __restrict__ xBC,
                         const float* __restrict__ dtbias,
                         const float* __restrict__ Alog,
                         const float* __restrict__ Dp,
                         float* __restrict__ yout) {
  const int p = threadIdx.x;           // 0..63
  const int head = blockIdx.x;         // 0..7
  const long b = blockIdx.y;           // 0..7
  const float A = -__expf(Alog[head]);
  const float Dv = Dp[head];
  const float dtb = dtbias[head];
  float h[DSTATE];
#pragma unroll
  for (int n = 0; n < DSTATE; ++n) h[n] = 0.f;
  __shared__ float sB[DSTATE], sC[DSTATE];
  for (int t = 0; t < TOK; ++t) {
    const long base = (b << 9) + t;
    if (p < 32) sB[p] = xBC[base * CDIM + DI + p];
    else        sC[p - 32] = xBC[base * CDIM + DI + DSTATE + (p - 32)];
    __syncthreads();
    const float dtv = softplusf(zx[base * (long)DPROJ + (DPROJ - NH) + head] + dtb);
    const float dA = __expf(dtv * A);
    const float xv = xBC[base * CDIM + head * HD + p];
    const float coef = dtv * xv;
    float y = 0.f;
#pragma unroll
    for (int n = 0; n < DSTATE; ++n) { h[n] = h[n] * dA + coef * sB[n]; y += h[n] * sC[n]; }
    yout[base * (long)DI + head * HD + p] = y + Dv * xv;
    __syncthreads();
  }
}

// gated rmsnorm over D=512, output bf16 (GEMM A operand)
__global__ void gatenorm512_bf(const float* __restrict__ y,
                               const float* __restrict__ zx,
                               const float* __restrict__ gw,
                               bf16_t* __restrict__ out) {
  __shared__ float red[256];
  const int row = blockIdx.x, c = threadIdx.x;
  const long zb = (long)row * DPROJ;
  const float g0 = y[(long)row * DI + c]       * siluf(zx[zb + c]);
  const float g1 = y[(long)row * DI + c + 256] * siluf(zx[zb + c + 256]);
  red[c] = g0 * g0 + g1 * g1;
  __syncthreads();
  for (int s = 128; s > 0; s >>= 1) { if (c < s) red[c] += red[c + s]; __syncthreads(); }
  const float scale = rsqrtf(red[0] * (1.f / DI) + 1e-5f);
  out[(long)row * DI + c]       = f2bf(g0 * scale * gw[c]);
  out[(long)row * DI + c + 256] = f2bf(g1 * scale * gw[c + 256]);
}

// concat [h_f, reverse(h_b)] -> bf16 (A of the patch_out GEMM)
__global__ void concat_bf(const float* __restrict__ hf, const float* __restrict__ hb,
                          bf16_t* __restrict__ out) {
  long idx = (long)blockIdx.x * 256 + threadIdx.x;   // ROWS * 512
  const int d = (int)(idx & 511);
  const long bt = idx >> 9;
  const int t = (int)(bt & 511);
  const long b = bt >> 9;
  const float v = (d < DM) ? hf[bt * DM + d]
                           : hb[((b << 9) + (TOK - 1 - t)) * DM + (d - DM)];
  out[idx] = f2bf(v);
}

// un-patch + slice to the generated region
__global__ void gather_out(const float* __restrict__ ppred, float* __restrict__ out) {
  const int idx = blockIdx.x * 256 + threadIdx.x;    // NB*OUTLEN*OUTC = 131072
  const int oc = idx & 7;
  const int tt = (idx >> 3) & (OUTLEN - 1);
  const int b  = idx >> 14;
  const int tau = TPRE + tt;
  const int t = tau >> 4, k = tau & 15;
  out[idx] = ppred[((long)(b << 9) + t) * (OUTC * 16) + oc * 16 + k];
}

// ============================================================================
extern "C" void kernel_launch(void* const* d_in, const int* in_sizes, int n_in,
                              void* d_out, int out_size, void* d_ws, size_t ws_size,
                              hipStream_t stream) {
  const float* x_prefix    = (const float*)d_in[0];
  const float* y_aux       = (const float*)d_in[1];
  const float* pre_conv_w  = (const float*)d_in[2];
  const float* pre_conv_b  = (const float*)d_in[3];
  const float* patch_w     = (const float*)d_in[4];
  const float* patch_b     = (const float*)d_in[5];
  const float* mask_token  = (const float*)d_in[6];
  const float* aux_w       = (const float*)d_in[7];
  const float* aux_b       = (const float*)d_in[8];
  const float* patch_out_w = (const float*)d_in[9];
  const float* patch_out_b = (const float*)d_in[10];
  const float* nw[2]   = { (const float*)d_in[11], (const float*)d_in[20] };
  const float* ip[2]   = { (const float*)d_in[12], (const float*)d_in[21] };
  const float* cw[2]   = { (const float*)d_in[13], (const float*)d_in[22] };
  const float* cb[2]   = { (const float*)d_in[14], (const float*)d_in[23] };
  const float* dtb[2]  = { (const float*)d_in[15], (const float*)d_in[24] };
  const float* alog[2] = { (const float*)d_in[16], (const float*)d_in[25] };
  const float* dpar[2] = { (const float*)d_in[17], (const float*)d_in[26] };
  const float* gn[2]   = { (const float*)d_in[18], (const float*)d_in[27] };
  const float* op[2]   = { (const float*)d_in[19], (const float*)d_in[28] };

  char* ws = (char*)d_ws;
  size_t off = 0;
  auto alloc = [&](size_t bytes) -> void* {
    void* p = ws + off;
    off += (bytes + 255) & ~(size_t)255;
    return p;
  };
  bf16_t* Apatch    = (bf16_t*)alloc((size_t)ROWS * RPATCH * 2);
  bf16_t* patchWbf  = (bf16_t*)alloc((size_t)DM * RPATCH * 2);
  bf16_t* inprojT   = (bf16_t*)alloc((size_t)4 * DPROJP * DM * 2);
  bf16_t* outprojT  = (bf16_t*)alloc((size_t)4 * DM * DI * 2);
  bf16_t* patchOutT = (bf16_t*)alloc((size_t)(OUTC * 16) * (2 * DM) * 2);
  float*  xtok      = (float*)alloc((size_t)ROWS * DM * 4);
  float*  auxbuf    = (float*)alloc((size_t)NB * DM * 4);
  float*  hbuf[2]   = { (float*)alloc((size_t)ROWS * DM * 4),
                        (float*)alloc((size_t)ROWS * DM * 4) };
  bf16_t* u_bf      = (bf16_t*)alloc((size_t)ROWS * DM * 2);
  float*  zx        = (float*)alloc((size_t)ROWS * DPROJ * 4);
  float*  xBC       = (float*)alloc((size_t)ROWS * CDIM * 4);
  float*  yout      = (float*)alloc((size_t)ROWS * DI * 4);
  bf16_t* yg        = (bf16_t*)alloc((size_t)ROWS * DI * 2);
  bf16_t* hcat      = (bf16_t*)alloc((size_t)ROWS * (2 * DM) * 2);
  float*  ppred     = (float*)alloc((size_t)ROWS * (OUTC * 16) * 4);

  // --- weight prep (bf16, transposed to N x K for WMMA B operand) ---
  cvt_bf<<<(DM * RPATCH + 255) / 256, 256, 0, stream>>>(patch_w, patchWbf, (long)DM * RPATCH);
  for (int d = 0; d < 2; ++d)
    for (int l = 0; l < 2; ++l) {
      cvtT_bf<<<(DPROJP * DM + 255) / 256, 256, 0, stream>>>(
          ip[d] + (long)l * DM * DPROJ, inprojT + (long)(d * 2 + l) * DPROJP * DM,
          DM, DPROJ, DPROJP);
      cvtT_bf<<<(DM * DI + 255) / 256, 256, 0, stream>>>(
          op[d] + (long)l * DI * DM, outprojT + (long)(d * 2 + l) * DM * DI,
          DI, DM, DM);
    }
  cvtT_bf<<<((OUTC * 16) * (2 * DM) + 255) / 256, 256, 0, stream>>>(
      patch_out_w, patchOutT, 2 * DM, OUTC * 16, OUTC * 16);

  // --- front end: fused pre-conv+SiLU -> patch GEMM -> token assembly ---
  build_patch_A<<<(long)ROWS * RPATCH / 256, 256, 0, stream>>>(
      x_prefix, pre_conv_w, pre_conv_b, Apatch);
  {
    dim3 g(DM / 64, ROWS / 16);
    gemm_bf16_wmma<<<g, 32, 0, stream>>>(Apatch, patchWbf, xtok,
                                         ROWS, DM, RPATCH, DM, nullptr, 0);
  }
  aux_bias_k<<<NB, DM, 0, stream>>>(y_aux, aux_w, aux_b, auxbuf);
  assemble_tokens<<<ROWS * DM / 256, 256, 0, stream>>>(
      xtok, patch_b, mask_token, auxbuf, hbuf[0], hbuf[1]);

  // --- bidirectional Mamba layers ---
  for (int d = 0; d < 2; ++d) {
    for (int l = 0; l < 2; ++l) {
      float* h = hbuf[d];
      const int wsel = d * 2 + l;
      rmsnorm256_bf<<<ROWS, 256, 0, stream>>>(h, nw[d] + l * DM, u_bf);
      {
        dim3 g(DPROJP / 64, ROWS / 16);
        gemm_bf16_wmma<<<g, 32, 0, stream>>>(u_bf, inprojT + (long)wsel * DPROJP * DM,
                                             zx, ROWS, DPROJ, DM, DPROJ, nullptr, 0);
      }
      dwconv_silu<<<ROWS * CDIM / 256, 256, 0, stream>>>(
          zx, cw[d] + l * CDIM * 4, cb[d] + l * CDIM, xBC);
      ssm_scan<<<dim3(NH, NB), HD, 0, stream>>>(
          zx, xBC, dtb[d] + l * NH, alog[d] + l * NH, dpar[d] + l * NH, yout);
      gatenorm512_bf<<<ROWS, 256, 0, stream>>>(yout, zx, gn[d] + l * DI, yg);
      {
        // residual fused: h += yg @ outproj  (accumulate=1)
        dim3 g(DM / 64, ROWS / 16);
        gemm_bf16_wmma<<<g, 32, 0, stream>>>(yg, outprojT + (long)wsel * DM * DI,
                                             h, ROWS, DM, DI, DM, nullptr, 1);
      }
    }
  }

  // --- head: concat -> patch_out GEMM (+bias) -> un-patch/slice ---
  concat_bf<<<ROWS * (2 * DM) / 256, 256, 0, stream>>>(hbuf[0], hbuf[1], hcat);
  {
    dim3 g((OUTC * 16) / 64, ROWS / 16);
    gemm_bf16_wmma<<<g, 32, 0, stream>>>(hcat, patchOutT, ppred,
                                         ROWS, OUTC * 16, 2 * DM, OUTC * 16,
                                         patch_out_b, 0);
  }
  gather_out<<<(NB * OUTLEN * OUTC) / 256, 256, 0, stream>>>(ppred, (float*)d_out);
}
